// TemporalAttentionLayer2_49624052138345
// MI455X (gfx1250) — compile-verified, matrix operand order checked by reading, not run on hardware
//
#include <hip/hip_runtime.h>

#define NE1   250000
#define NETOT 550000        // 2*NE1 + NND
#define NND   50000
#define ATT_SCALE 0.125f    // (OUT_DIM/N_HEAD)^-0.5 = 64^-0.5

typedef __attribute__((ext_vector_type(16))) __bf16 v16bf;
typedef __attribute__((ext_vector_type(8)))  __bf16 v8bf;
typedef __attribute__((ext_vector_type(8)))  float  v8f;

// ---------------------------------------------------------------- helpers ---

// Augmented edge list: [fwd edges | reversed edges | self loops]
__device__ inline void edge_decode(int e, const int* ei, int& src, int& tgt,
                                   int& fe, bool& self) {
    if (e < NE1)            { src = ei[2*e];        tgt = ei[2*e+1];        fe = e;  self = false; }
    else if (e < 2*NE1)     { fe = e - NE1;         src = ei[2*fe+1];       tgt = ei[2*fe]; self = false; }
    else                    { src = tgt = e - 2*NE1; fe = 0;                self = true; }
}

__device__ inline int edge_src(int e, const int* ei) {
    if (e < NE1)        return ei[2*e];
    if (e < 2*NE1)      return ei[2*(e-NE1)+1];
    return e - 2*NE1;
}

// A-fragment: 16x32 bf16 tile, row-major in LDS with leading dim `ld`.
// Lane L: row = L&15, kb = (L<16)?0:8 ; a[j] = A[row][k0+kb+j] (j<8),
//                                       a[8+j] = A[row][k0+kb+16+j]
__device__ inline v16bf load_frag_A(const __bf16* base, int ld, int lane, int k0) {
    int row = lane & 15;
    int kb  = (lane < 16) ? 0 : 8;
    const __bf16* p = base + row * ld + k0 + kb;
    v16bf a;
#pragma unroll
    for (int j = 0; j < 8; ++j) a[j] = p[j];
#pragma unroll
    for (int j = 0; j < 8; ++j) a[8 + j] = p[16 + j];
    return a;
}

// B-fragment from pre-swizzled weights: Wsw[((kt*8+nt)*32 + lane)*16 + j]
// One aligned 32-byte load per lane (-> 2x global_load_b128).
__device__ inline v16bf load_frag_Bsw(const __bf16* __restrict__ Wsw,
                                      int lane, int kt, int nt) {
    return *(const v16bf*)(Wsw + (size_t)(((kt * 8 + nt) * 32) + lane) * 16);
}

__device__ inline v8f wmma_bf16(v16bf a, v16bf b, v8f c) {
    return __builtin_amdgcn_wmma_f32_16x16x32_bf16(false, a, false, b,
                                                   (short)0, c, false, false);
}

// GEMM for one 16xK A-tile (in LDS) against one 16-wide N-tile of a swizzled
// weight. B-fragment stream is software-pipelined: load of fragment kt+1 is
// issued before the WMMA consuming fragment kt, so each s_wait covers a load
// that has been in flight for a full WMMA.
template <int KT>
__device__ inline v8f gemm_tile(const __bf16* sA, int ld,
                                const __bf16* __restrict__ Wsw,
                                int lane, int nt) {
    v16bf a[KT];
#pragma unroll
    for (int kt = 0; kt < KT; ++kt) a[kt] = load_frag_A(sA, ld, lane, kt * 32);
    v8f acc = {};
    v16bf b = load_frag_Bsw(Wsw, lane, 0, nt);
#pragma unroll
    for (int kt = 0; kt < KT; ++kt) {
        v16bf bn = (kt + 1 < KT) ? load_frag_Bsw(Wsw, lane, kt + 1, nt) : b;
        acc = wmma_bf16(a[kt], b, acc);
        b = bn;
    }
    return acc;
}

// Swizzle one element of a row-major [K][128] f32 weight into fragment-major bf16.
__device__ inline void swizzle_one(const float* __restrict__ W,
                                   __bf16* __restrict__ Wsw, int i) {
    int j  = i & 15;            // element within lane fragment
    int L  = (i >> 4) & 31;     // lane
    int t  = i >> 9;            // tile = kt*8 + nt
    int kt = t >> 3, nt = t & 7;
    int col  = nt * 16 + (L & 15);
    int koff = ((L < 16) ? 0 : 8) + j + ((j >= 8) ? 8 : 0);
    Wsw[i] = (__bf16)W[(kt * 32 + koff) * 128 + col];
}

// ------------------------------------------------------------------ prep ----

__global__ void prep_kernel(const float* Wq, const float* Wk, const float* Wv,
                            const float* W1, const float* W2,
                            __bf16* Wq_sw, __bf16* Wk_sw, __bf16* Wv_sw,
                            __bf16* W1_sw, __bf16* W2_sw,
                            float* denom, unsigned* mbuf, float* out_emb) {
    int i = blockIdx.x * blockDim.x + threadIdx.x;
    if (i < 192*128) swizzle_one(Wq, Wq_sw, i);
    if (i < 256*128) { swizzle_one(Wk, Wk_sw, i);
                       swizzle_one(Wv, Wv_sw, i);
                       swizzle_one(W1, W1_sw, i); }
    if (i < 128*128) swizzle_one(W2, W2_sw, i);
    if (i < NND*2)   { denom[i] = 0.0f; mbuf[i] = 0u; }
    if (i < NND*128) out_emb[i] = 0.0f;
}

// --------------------------------------------- edge QKV + attention logits --

__global__ __launch_bounds__(32)
void edge_qkv_att(const float* __restrict__ nodef, const float* __restrict__ ef,
                  const float* __restrict__ stf,   const float* __restrict__ et,
                  const __bf16* __restrict__ Wq, const float* __restrict__ bq,
                  const __bf16* __restrict__ Wk, const float* __restrict__ bk,
                  const __bf16* __restrict__ Wv, const float* __restrict__ bv,
                  const int* __restrict__ ei,
                  __bf16* __restrict__ vbuf, float* __restrict__ attbuf,
                  unsigned* __restrict__ mbuf) {
    __shared__ __bf16 sSrc[16 * 192];   // [edge][128 nodef | 64 stf]
    __shared__ __bf16 sTgt[16 * 256];   // [edge][128 nodef | 64 ef | 64 et]
    __shared__ float  sQ[16 * 128];
    __shared__ float  sK[16 * 128];

    const int lane = threadIdx.x;
    const int e0   = blockIdx.x * 16;   // 550000/16 = 34375 tiles, exact

    // ---- gather + f32->bf16 into LDS
    for (int r = 0; r < 16; ++r) {
        int e = e0 + r, src, tgt, fe; bool self;
        edge_decode(e, ei, src, tgt, fe, self);
        for (int c = lane; c < 192; c += 32) {
            float x = (c < 128) ? nodef[src*128 + c]
                                : (self ? stf[c-128] : stf[fe*64 + (c-128)]);
            sSrc[r*192 + c] = (__bf16)x;
        }
        for (int c = lane; c < 256; c += 32) {
            float x;
            if (c < 128)        x = nodef[tgt*128 + c];
            else if (c < 192)   x = self ? 0.0f : ef[fe*64 + (c-128)];
            else                x = self ? stf[c-192] : et[fe*64 + (c-192)];
            sTgt[r*256 + c] = (__bf16)x;
        }
    }
    __syncthreads();   // single-wave WG: acts as LDS fence

    const int n_lo = lane & 15;
    const int mB   = (lane < 16) ? 0 : 8;

    // hoist per-lane bias values (one clause, no per-tile load stalls)
    float bqv[8], bkv[8], bvv[8];
#pragma unroll
    for (int nt = 0; nt < 8; ++nt) {
        bqv[nt] = bq[nt*16 + n_lo];
        bkv[nt] = bk[nt*16 + n_lo];
        bvv[nt] = bv[nt*16 + n_lo];
    }

    // ---- Q = relu(src @ Wq + bq)    (K = 192)
    for (int nt = 0; nt < 8; ++nt) {
        v8f acc = gemm_tile<6>(sSrc, 192, Wq, lane, nt);
        int n = nt*16 + n_lo;
#pragma unroll
        for (int r = 0; r < 8; ++r) {
            float v = acc[r] + bqv[nt];
            sQ[(mB + r)*128 + n] = v > 0.0f ? v : 0.0f;
        }
    }

    // ---- K = relu(tgt @ Wk + bk)    (K = 256)
    for (int nt = 0; nt < 8; ++nt) {
        v8f acc = gemm_tile<8>(sTgt, 256, Wk, lane, nt);
        int n = nt*16 + n_lo;
#pragma unroll
        for (int r = 0; r < 8; ++r) {
            float v = acc[r] + bkv[nt];
            sK[(mB + r)*128 + n] = v > 0.0f ? v : 0.0f;
        }
    }

    // ---- V = tgt @ Wv + bv -> global, stored in C-fragment layout:
    //      vbuf[(((tile*8 + nt)*32) + lane)*8 + r]  (one 16B store per nt)
    for (int nt = 0; nt < 8; ++nt) {
        v8f acc = gemm_tile<8>(sTgt, 256, Wv, lane, nt);
        v8bf pv;
#pragma unroll
        for (int r = 0; r < 8; ++r) pv[r] = (__bf16)(acc[r] + bvv[nt]);
        *(v8bf*)(vbuf + (size_t)(((size_t)blockIdx.x * 8 + nt) * 32 + lane) * 8) = pv;
    }
    __syncthreads();

    // ---- logits: lane = (edge 0..15, head 0..1); q,k >= 0 so att >= 0
    const int eL = lane & 15, h = lane >> 4;
    float s = 0.0f;
    for (int d = 0; d < 64; ++d)
        s += sQ[eL*128 + h*64 + d] * sK[eL*128 + h*64 + d];
    s *= ATT_SCALE;
    int e   = e0 + eL;
    int src = edge_src(e, ei);
    attbuf[e*2 + h] = s;
    atomicMax(&mbuf[src*2 + h], __float_as_uint(s));  // valid: s >= 0
}

// ----------------------------------------------------- softmax pass 2 -------

__global__ void exp_denom_kernel(const int* __restrict__ ei,
                                 float* __restrict__ attbuf,
                                 const unsigned* __restrict__ mbuf,
                                 float* __restrict__ denom) {
    int idx = blockIdx.x * blockDim.x + threadIdx.x;
    if (idx >= NETOT * 2) return;
    int e = idx >> 1, h = idx & 1;
    int src = edge_src(e, ei);
    float m  = __uint_as_float(mbuf[src*2 + h]);
    float ev = __expf(attbuf[idx] - m);
    attbuf[idx] = ev;
    atomicAdd(&denom[src*2 + h], ev);
}

// ----------------------------------------------------- message scatter ------

__global__ void message_kernel(const int* __restrict__ ei,
                               const float* __restrict__ attbuf,
                               const float* __restrict__ denom,
                               const __bf16* __restrict__ vbuf,
                               float* __restrict__ out_emb) {
    long idx = (long)blockIdx.x * blockDim.x + threadIdx.x;
    if (idx >= (long)NETOT * 128) return;
    int e = (int)(idx >> 7), d = (int)(idx & 127), h = d >> 6;
    int src = edge_src(e, ei);
    float score = attbuf[e*2 + h] / denom[src*2 + h];
    // V is stored in C-fragment layout
    int tile = e >> 4, m = e & 15;
    int nt = d >> 4, nl = d & 15;
    int lane = nl + ((m & 8) ? 16 : 0);
    int r = m & 7;
    size_t vIdx = ((size_t)(tile * 8 + nt) * 32 + lane) * 8 + r;
    atomicAdd(&out_emb[src*128 + d], score * (float)vbuf[vIdx]);
}

// ----------------------------------------------------- node MLP (WMMA) ------

__global__ __launch_bounds__(32)
void node_ffn(const float* __restrict__ out_emb, const float* __restrict__ nodef,
              const __bf16* __restrict__ W1, const float* __restrict__ b1,
              const __bf16* __restrict__ W2, const float* __restrict__ b2,
              float* __restrict__ out) {
    __shared__ __bf16 sX[16 * 256];   // [node][128 out_emb | 128 nodef]
    __shared__ __bf16 sH[16 * 128];

    const int lane = threadIdx.x;
    const int nd0  = blockIdx.x * 16;  // 50000/16 = 3125 tiles, exact

    for (int r = 0; r < 16; ++r) {
        int node = nd0 + r;
        for (int c = lane; c < 256; c += 32) {
            float x = (c < 128) ? out_emb[node*128 + c] : nodef[node*128 + (c-128)];
            sX[r*256 + c] = (__bf16)x;
        }
    }
    __syncthreads();

    const int n_lo = lane & 15;
    const int mB   = (lane < 16) ? 0 : 8;

    float b1v[8], b2v[8];
#pragma unroll
    for (int nt = 0; nt < 8; ++nt) {
        b1v[nt] = b1[nt*16 + n_lo];
        b2v[nt] = b2[nt*16 + n_lo];
    }

    // h = relu(X @ W1 + b1)   K = 256
    for (int nt = 0; nt < 8; ++nt) {
        v8f acc = gemm_tile<8>(sX, 256, W1, lane, nt);
        int n = nt*16 + n_lo;
#pragma unroll
        for (int r = 0; r < 8; ++r) {
            float v = acc[r] + b1v[nt];
            sH[(mB + r)*128 + n] = (__bf16)(v > 0.0f ? v : 0.0f);
        }
    }
    __syncthreads();

    // out = h @ W2 + b2       K = 128
    for (int nt = 0; nt < 8; ++nt) {
        v8f acc = gemm_tile<4>(sH, 128, W2, lane, nt);
        int n = nt*16 + n_lo;
#pragma unroll
        for (int r = 0; r < 8; ++r)
            out[(long)(nd0 + mB + r)*128 + n] = acc[r] + b2v[nt];
    }
}

// ------------------------------------------------------------- launcher -----

extern "C" void kernel_launch(void* const* d_in, const int* in_sizes, int n_in,
                              void* d_out, int out_size, void* d_ws, size_t ws_size,
                              hipStream_t stream) {
    const float* nodef = (const float*)d_in[0];
    const float* ef    = (const float*)d_in[1];
    const float* stf   = (const float*)d_in[2];
    const float* et    = (const float*)d_in[3];
    const float* Wq    = (const float*)d_in[4];
    const float* bq    = (const float*)d_in[5];
    const float* Wk    = (const float*)d_in[6];
    const float* bk    = (const float*)d_in[7];
    const float* Wv    = (const float*)d_in[8];
    const float* bv    = (const float*)d_in[9];
    const float* W1    = (const float*)d_in[10];
    const float* b1    = (const float*)d_in[11];
    const float* W2    = (const float*)d_in[12];
    const float* b2    = (const float*)d_in[13];
    const int*   ei    = (const int*)d_in[14];
    float* out = (float*)d_out;

    // ---- workspace carve-up (256B aligned)
    char* ws = (char*)d_ws;
    size_t off = 0;
    auto alloc = [&](size_t bytes) {
        void* p = ws + off;
        off = (off + bytes + 255) & ~(size_t)255;
        return p;
    };
    __bf16*   Wq_sw   = (__bf16*)  alloc(192*128*sizeof(__bf16));
    __bf16*   Wk_sw   = (__bf16*)  alloc(256*128*sizeof(__bf16));
    __bf16*   Wv_sw   = (__bf16*)  alloc(256*128*sizeof(__bf16));
    __bf16*   W1_sw   = (__bf16*)  alloc(256*128*sizeof(__bf16));
    __bf16*   W2_sw   = (__bf16*)  alloc(128*128*sizeof(__bf16));
    float*    attbuf  = (float*)   alloc((size_t)NETOT*2*sizeof(float));
    unsigned* mbuf    = (unsigned*)alloc((size_t)NND*2*sizeof(unsigned));
    float*    denom   = (float*)   alloc((size_t)NND*2*sizeof(float));
    float*    out_emb = (float*)   alloc((size_t)NND*128*sizeof(float));
    __bf16*   vbuf    = (__bf16*)  alloc((size_t)NETOT*128*sizeof(__bf16));
    (void)ws_size; (void)n_in; (void)in_sizes; (void)out_size;

    // 1) weight swizzle (f32 -> fragment-major bf16) + zero init
    prep_kernel<<<(NND*128 + 255)/256, 256, 0, stream>>>(
        Wq, Wk, Wv, W1, W2, Wq_sw, Wk_sw, Wv_sw, W1_sw, W2_sw,
        denom, mbuf, out_emb);

    // 2) edge QKV + logits + segment max (34375 tiles of 16 edges, 1 wave each)
    edge_qkv_att<<<NETOT/16, 32, 0, stream>>>(
        nodef, ef, stf, et, Wq_sw, bq, Wk_sw, bk, Wv_sw, bv, ei,
        vbuf, attbuf, mbuf);

    // 3) exp + denominators
    exp_denom_kernel<<<(NETOT*2 + 255)/256, 256, 0, stream>>>(
        ei, attbuf, mbuf, denom);

    // 4) weighted scatter of V into out_emb
    long msg_items = (long)NETOT * 128;
    message_kernel<<<(unsigned)((msg_items + 255)/256), 256, 0, stream>>>(
        ei, attbuf, denom, vbuf, out_emb);

    // 5) node MLP (3125 tiles of 16 nodes, 1 wave each)
    node_ffn<<<NND/16, 32, 0, stream>>>(
        out_emb, nodef, W1_sw, b1, W2_sw, b2, out);
}
